// MultiHeadAttention_89601607729652
// MI455X (gfx1250) — matrix operand order repaired and checked by reading, not compiled, not run on hardware
//
#include <hip/hip_runtime.h>
#include <hip/hip_bf16.h>
#include <math.h>

typedef __attribute__((ext_vector_type(16))) _Float16 v16h;
typedef __attribute__((ext_vector_type(8)))  _Float16 v8h;
typedef __attribute__((ext_vector_type(8)))  float    v8f;

#define B_  8
#define S_  1024
#define D_  128
#define H_  8
#define HD_ 1024
#define M_TOTAL (B_ * S_)   // 8192

__device__ __forceinline__ v16h hcombine(v8h lo, v8h hi) {
  v16h r;
#pragma unroll
  for (int i = 0; i < 8; ++i) { r[i] = lo[i]; r[i + 8] = hi[i]; }
  return r;
}

__device__ __forceinline__ v8f wmma_f16(v16h a, v16h b, v8f c) {
  // D = A(16x32 f16) * B(32x16 f16) + C(16x16 f32)
  return __builtin_amdgcn_wmma_f32_16x16x32_f16(false, a, false, b,
                                                (short)0, c, false, false);
}

// CDNA5 async global->LDS copy (ASYNCcnt-tracked, bypasses VGPRs).
// VDST operand = 32-bit LDS byte offset (low 32 bits of the generic pointer
// to a __shared__ object are the LDS offset; HW adds LDS_BASE).
__device__ __forceinline__ void async_b128(const _Float16* g, _Float16* l) {
  const uint32_t lds = (uint32_t)(uintptr_t)l;
  asm volatile("global_load_async_to_lds_b128 %0, %1, off"
               :
               : "v"(lds), "v"(g)
               : "memory");
}

__device__ __forceinline__ void wait_async0() {
#if __has_builtin(__builtin_amdgcn_s_wait_asynccnt)
  __builtin_amdgcn_s_wait_asynccnt(0);
#else
  asm volatile("s_wait_asynccnt 0" ::: "memory");
#endif
}

// ---------------------------------------------------------------------------
// Kernel 1: q/k/v = (X + pos) @ W + b   (M=8192, K=128, N=1024), f16 output.
// blockIdx.z selects which of the three projections. 128x128 tile per block.
// ---------------------------------------------------------------------------
__global__ __launch_bounds__(256) void qkv_proj_kernel(
    const float* __restrict__ query, const float* __restrict__ key_in,
    const float* __restrict__ value, const float* __restrict__ pos,
    const float* __restrict__ Wq, const float* __restrict__ bq,
    const float* __restrict__ Wk, const float* __restrict__ bk,
    const float* __restrict__ Wv, const float* __restrict__ bv,
    _Float16* __restrict__ q_ws, _Float16* __restrict__ k_ws,
    _Float16* __restrict__ v_ws)
{
  const int which = blockIdx.z;
  const float* X    = (which == 0) ? query : (which == 1) ? key_in : value;
  const float* W    = (which == 0) ? Wq    : (which == 1) ? Wk     : Wv;
  const float* bias = (which == 0) ? bq    : (which == 1) ? bk     : bv;
  _Float16*    Y    = (which == 0) ? q_ws  : (which == 1) ? k_ws   : v_ws;

  const int mt  = blockIdx.x;   // 64 tiles of 128 rows
  const int nt  = blockIdx.y;   // 8 tiles of 128 cols
  const int tid = threadIdx.x;
  const int wave = tid >> 5, lane = tid & 31, l16 = lane & 15, g = lane >> 4;

  __shared__ __align__(16) _Float16 Atile[128 * 128];   // [m][k]
  __shared__ __align__(16) _Float16 BtileT[128 * 128];  // [n][k]

  // Stage A = (X + pos), fp32 -> fp16 (16 float4 per thread)
#pragma unroll
  for (int it = 0; it < 16; ++it) {
    int o = (tid + it * 256) * 4;
    int r = o >> 7, c = o & 127;
    size_t src = (size_t)(mt * 128 + r) * D_ + c;
    float4 xv = *(const float4*)&X[src];
    float4 pv = *(const float4*)&pos[src];
    _Float16* dst = &Atile[r * 128 + c];
    dst[0] = (_Float16)(xv.x + pv.x);
    dst[1] = (_Float16)(xv.y + pv.y);
    dst[2] = (_Float16)(xv.z + pv.z);
    dst[3] = (_Float16)(xv.w + pv.w);
  }
  // Stage W tile transposed so B-fragment reads are contiguous in K
#pragma unroll
  for (int it = 0; it < 16; ++it) {
    int o = (tid + it * 256) * 4;
    int k = o >> 7, c = o & 127;
    float4 wv = *(const float4*)&W[(size_t)k * HD_ + nt * 128 + c];
    BtileT[(c + 0) * 128 + k] = (_Float16)wv.x;
    BtileT[(c + 1) * 128 + k] = (_Float16)wv.y;
    BtileT[(c + 2) * 128 + k] = (_Float16)wv.z;
    BtileT[(c + 3) * 128 + k] = (_Float16)wv.w;
  }
  __syncthreads();

  // A fragments for this wave's 16 rows, full K=128 (4 chunks of 32)
  v16h a[4];
#pragma unroll
  for (int kc = 0; kc < 4; ++kc) {
    const int row = wave * 16 + l16;
    v8h lo = *(const v8h*)&Atile[row * 128 + kc * 32 + g * 8];
    v8h hi = *(const v8h*)&Atile[row * 128 + kc * 32 + 16 + g * 8];
    a[kc] = hcombine(lo, hi);
  }

#pragma unroll
  for (int ntile = 0; ntile < 8; ++ntile) {
    v8f c = {};
#pragma unroll
    for (int kc = 0; kc < 4; ++kc) {
      const int col = ntile * 16 + l16;
      v8h lo = *(const v8h*)&BtileT[col * 128 + kc * 32 + g * 8];
      v8h hi = *(const v8h*)&BtileT[col * 128 + kc * 32 + 16 + g * 8];
      c = wmma_f16(a[kc], hcombine(lo, hi), c);
    }
    const int coln = nt * 128 + ntile * 16 + l16;
    const float bb = bias[coln];
#pragma unroll
    for (int r = 0; r < 8; ++r) {
      int m = mt * 128 + wave * 16 + r + 8 * g;
      Y[(size_t)m * HD_ + coln] = (_Float16)(c[r] + bb);
    }
  }
}

// ---------------------------------------------------------------------------
// Kernel 2: flash attention per (b, h). 128 query rows per block (16 per
// wave). K/V double-buffered through LDS in 32-key chunks; K staged with
// async global->LDS copies overlapped with WMMA compute; online softmax.
// ---------------------------------------------------------------------------
__global__ __launch_bounds__(256) void attn_kernel(
    const _Float16* __restrict__ q_ws, const _Float16* __restrict__ k_ws,
    const _Float16* __restrict__ v_ws, const int* __restrict__ mask,
    _Float16* __restrict__ attn_ws)
{
  const int qb = blockIdx.x;   // 0..7 (128 q rows each)
  const int h  = blockIdx.y;
  const int b  = blockIdx.z;
  const int tid = threadIdx.x, wave = tid >> 5, lane = tid & 31;
  const int l16 = lane & 15, g = lane >> 4;

  __shared__ __align__(16) _Float16 K_lds[2][32 * 128];    // [key][d]
  __shared__ __align__(16) _Float16 V_ldsT[2][128 * 32];   // [d][key]
  __shared__ __align__(16) _Float16 P_lds[8][16 * 32];     // per-wave scratch

  // Q fragments (A-layout), resident in VGPRs for the whole kernel
  const int qrow = qb * 128 + wave * 16 + l16;
  const size_t qbase = ((size_t)(b * S_ + qrow)) * HD_ + h * D_;
  v16h aq[4];
#pragma unroll
  for (int kc = 0; kc < 4; ++kc) {
    v8h lo = *(const v8h*)&q_ws[qbase + kc * 32 + g * 8];
    v8h hi = *(const v8h*)&q_ws[qbase + kc * 32 + 16 + g * 8];
    aq[kc] = hcombine(lo, hi);
  }

  v8f acc[8] = {};
  float mrow[8], lrow[8];
#pragma unroll
  for (int r = 0; r < 8; ++r) { mrow[r] = -1e30f; lrow[r] = 0.0f; }

  const float scale = 0.08838834764831845f;  // 1/sqrt(128)

  const int so = tid * 16;           // staging: 16 halves per thread
  const int sr = so >> 7, sc = so & 127;

  // Stage one 32-key chunk of K (async -> LDS) and V (transposed via VGPRs)
  auto stage = [&](int kt, int buf) {
    const size_t src = ((size_t)(b * S_ + kt * 32 + sr)) * HD_ + h * D_ + sc;
    async_b128(&k_ws[src],     &K_lds[buf][sr * 128 + sc]);
    async_b128(&k_ws[src + 8], &K_lds[buf][sr * 128 + sc + 8]);
    v8h v0 = *(const v8h*)&v_ws[src];
    v8h v1 = *(const v8h*)&v_ws[src + 8];
#pragma unroll
    for (int i = 0; i < 8; ++i) {
      V_ldsT[buf][(sc + i) * 32 + sr]     = v0[i];
      V_ldsT[buf][(sc + 8 + i) * 32 + sr] = v1[i];
    }
  };

  stage(0, 0);
  wait_async0();
  __syncthreads();

  for (int kt = 0; kt < 32; ++kt) {
    const int buf = kt & 1;
    // Issue next chunk's copies before computing on the current one.
    if (kt + 1 < 32) stage(kt + 1, buf ^ 1);

    // ---- scores for 32 keys: two 16x16 C tiles, K-dim = D = 128 ----
    v8f s[2];
#pragma unroll
    for (int hf = 0; hf < 2; ++hf) {
      v8f c = {};
#pragma unroll
      for (int kc = 0; kc < 4; ++kc) {
        const int krow = hf * 16 + l16;
        v8h lo = *(const v8h*)&K_lds[buf][krow * 128 + kc * 32 + g * 8];
        v8h hi = *(const v8h*)&K_lds[buf][krow * 128 + kc * 32 + 16 + g * 8];
        c = wmma_f16(aq[kc], hcombine(lo, hi), c);
      }
      const int key = kt * 32 + hf * 16 + l16;
      const float madd = mask[b * S_ + key] ? 0.0f : -1e30f;
#pragma unroll
      for (int r = 0; r < 8; ++r) c[r] = c[r] * scale + madd;
      s[hf] = c;
    }

    // ---- online softmax (row = r + 8*g lives in a 16-lane group) ----
#pragma unroll
    for (int r = 0; r < 8; ++r) {
      float mx = fmaxf(s[0][r], s[1][r]);
      mx = fmaxf(mx, __shfl_xor(mx, 1, 32));
      mx = fmaxf(mx, __shfl_xor(mx, 2, 32));
      mx = fmaxf(mx, __shfl_xor(mx, 4, 32));
      mx = fmaxf(mx, __shfl_xor(mx, 8, 32));
      const float mnew = fmaxf(mrow[r], mx);
      const float al = __expf(mrow[r] - mnew);
      mrow[r] = mnew;
      const float p0 = __expf(s[0][r] - mnew);
      const float p1 = __expf(s[1][r] - mnew);
      s[0][r] = p0; s[1][r] = p1;
      float rs = p0 + p1;
      rs += __shfl_xor(rs, 1, 32);
      rs += __shfl_xor(rs, 2, 32);
      rs += __shfl_xor(rs, 4, 32);
      rs += __shfl_xor(rs, 8, 32);
      lrow[r] = lrow[r] * al + rs;
#pragma unroll
      for (int dt = 0; dt < 8; ++dt) acc[dt][r] *= al;
    }

    // ---- P: C-layout f32 -> A-layout f16 via per-wave LDS scratch ----
#pragma unroll
    for (int r = 0; r < 8; ++r) {
      const int row = r + 8 * g;
      P_lds[wave][row * 32 + l16]      = (_Float16)s[0][r];
      P_lds[wave][row * 32 + 16 + l16] = (_Float16)s[1][r];
    }
    v8h plo = *(const v8h*)&P_lds[wave][l16 * 32 + g * 8];
    v8h phi = *(const v8h*)&P_lds[wave][l16 * 32 + 16 + g * 8];
    v16h ap = hcombine(plo, phi);

    // ---- out += P @ V  (contraction over 32 keys) ----
#pragma unroll
    for (int dt = 0; dt < 8; ++dt) {
      v8h lo = *(const v8h*)&V_ldsT[buf][(dt * 16 + l16) * 32 + g * 8];
      v8h hi = *(const v8h*)&V_ldsT[buf][(dt * 16 + l16) * 32 + 16 + g * 8];
      acc[dt] = wmma_f16(ap, hcombine(lo, hi), acc[dt]);
    }

    // Next buffer's async copies must have landed before anyone reads it.
    wait_async0();
    __syncthreads();
  }

  // ---- normalize + store to (b, s, h*D + d) f16 for the final GEMM ----
#pragma unroll
  for (int dt = 0; dt < 8; ++dt) {
#pragma unroll
    for (int r = 0; r < 8; ++r) {
      const int srow = qb * 128 + wave * 16 + r + 8 * g;
      const float ov = acc[dt][r] / lrow[r];
      attn_ws[((size_t)(b * S_ + srow)) * HD_ + h * D_ + dt * 16 + l16] =
          (_Float16)ov;
    }
  }
}

// ---------------------------------------------------------------------------
// Kernel 3: out = attn @ Wo + bo   (M=8192, K=1024, N=128), f32 output.
// A tile staged with async global->LDS copies (already f16 in workspace).
// ---------------------------------------------------------------------------
__global__ __launch_bounds__(256) void out_proj_kernel(
    const _Float16* __restrict__ attn_ws, const float* __restrict__ Wo,
    const float* __restrict__ bo, float* __restrict__ out)
{
  const int mt = blockIdx.x;   // 64 tiles of 128 rows
  const int tid = threadIdx.x, wave = tid >> 5, lane = tid & 31;
  const int l16 = lane & 15, g = lane >> 4;

  __shared__ __align__(16) _Float16 Atile[128 * 128];  // [m][k]
  __shared__ __align__(16) _Float16 WoT[128 * 128];    // [n][k]

  v8f acc[8] = {};

  for (int kc = 0; kc < 8; ++kc) {
    // stage A chunk (already f16): async global->LDS b128 copies
#pragma unroll
    for (int it = 0; it < 8; ++it) {
      int o = tid * 64 + it * 8;
      int r = o >> 7, c = o & 127;
      async_b128(&attn_ws[(size_t)(mt * 128 + r) * HD_ + kc * 128 + c],
                 &Atile[r * 128 + c]);
    }
    // stage Wo chunk transposed, f32 -> f16
#pragma unroll
    for (int it = 0; it < 16; ++it) {
      int o = (tid + it * 256) * 4;
      int kl = o >> 7, c = o & 127;
      float4 wv = *(const float4*)&Wo[(size_t)(kc * 128 + kl) * D_ + c];
      WoT[(c + 0) * 128 + kl] = (_Float16)wv.x;
      WoT[(c + 1) * 128 + kl] = (_Float16)wv.y;
      WoT[(c + 2) * 128 + kl] = (_Float16)wv.z;
      WoT[(c + 3) * 128 + kl] = (_Float16)wv.w;
    }
    if (kc + 1 < 8)  // global_prefetch_b8 next Wo chunk
      __builtin_prefetch(&Wo[(size_t)((kc + 1) * 128) * D_ + tid * 4], 0, 1);
    wait_async0();
    __syncthreads();

    v16h a[4];
#pragma unroll
    for (int s = 0; s < 4; ++s) {
      const int row = wave * 16 + l16;
      v8h lo = *(const v8h*)&Atile[row * 128 + s * 32 + g * 8];
      v8h hi = *(const v8h*)&Atile[row * 128 + s * 32 + 16 + g * 8];
      a[s] = hcombine(lo, hi);
    }
#pragma unroll
    for (int ntile = 0; ntile < 8; ++ntile) {
#pragma unroll
      for (int s = 0; s < 4; ++s) {
        v8h lo = *(const v8h*)&WoT[(ntile * 16 + l16) * 128 + s * 32 + g * 8];
        v8h hi = *(const v8h*)&WoT[(ntile * 16 + l16) * 128 + s * 32 + 16 + g * 8];
        acc[ntile] = wmma_f16(a[s], hcombine(lo, hi), acc[ntile]);
      }
    }
    __syncthreads();
  }

#pragma unroll
  for (int ntile = 0; ntile < 8; ++ntile) {
    const int col = ntile * 16 + l16;
    const float bb = bo[col];
#pragma unroll
    for (int r = 0; r < 8; ++r) {
      const int m = mt * 128 + wave * 16 + r + 8 * g;
      out[(size_t)m * D_ + col] = acc[ntile][r] + bb;
    }
  }
}

// ---------------------------------------------------------------------------
extern "C" void kernel_launch(void* const* d_in, const int* in_sizes, int n_in,
                              void* d_out, int out_size, void* d_ws, size_t ws_size,
                              hipStream_t stream)
{
  const float* query  = (const float*)d_in[0];
  const float* key_in = (const float*)d_in[1];
  const float* value  = (const float*)d_in[2];
  const float* pos    = (const float*)d_in[3];
  const int*   mask   = (const int*)d_in[4];
  const float* Wq     = (const float*)d_in[5];
  const float* bq     = (const float*)d_in[6];
  const float* Wk     = (const float*)d_in[7];
  const float* bk     = (const float*)d_in[8];
  const float* Wv     = (const float*)d_in[9];
  const float* bv     = (const float*)d_in[10];
  const float* Wo     = (const float*)d_in[11];
  const float* bo     = (const float*)d_in[12];
  float* out = (float*)d_out;

  const size_t proj_elems = (size_t)M_TOTAL * HD_;   // 8192*1024 halves each
  _Float16* q_ws    = (_Float16*)d_ws;
  _Float16* k_ws    = q_ws + proj_elems;
  _Float16* v_ws    = k_ws + proj_elems;
  _Float16* attn_ws = v_ws + proj_elems;             // 64 MB total workspace

  dim3 blk(256, 1, 1);
  qkv_proj_kernel<<<dim3(64, 8, 3), blk, 0, stream>>>(
      query, key_in, value, pos, Wq, bq, Wk, bk, Wv, bv, q_ws, k_ws, v_ws);
  attn_kernel<<<dim3(8, 8, 8), blk, 0, stream>>>(q_ws, k_ws, v_ws, mask, attn_ws);
  out_proj_kernel<<<dim3(64, 1, 1), blk, 0, stream>>>(attn_ws, Wo, bo, out);
}
